// BiLSTM_Fuzzy_CRF_63891933495355
// MI455X (gfx1250) — matrix operand order repaired and checked by reading, not compiled, not run on hardware
//
#include <hip/hip_runtime.h>
#include <hip/hip_bf16.h>

typedef __attribute__((ext_vector_type(16))) _Float16 v16h;
typedef __attribute__((ext_vector_type(8)))  float    v8f;

#define B_  256
#define S_  1024
#define T_  64
#define NEGV  (-10000000.0f)
#define LOG2E (1.4426950408889634f)
#define LN2   (0.6931471805599453f)

// raw-hardware transcendentals: args are well-conditioned (no denorm/inf fixups needed)
__device__ __forceinline__ float fast_exp(float x) {
#if __has_builtin(__builtin_amdgcn_exp2f)
    return __builtin_amdgcn_exp2f(x * LOG2E);   // v_exp_f32
#else
    return __expf(x);
#endif
}
__device__ __forceinline__ float fast_log(float x) {
#if __has_builtin(__builtin_amdgcn_logf)
    return __builtin_amdgcn_logf(x) * LN2;      // v_log_f32
#else
    return __logf(x);
#endif
}

// streaming logsumexp accumulator
__device__ __forceinline__ void lse_acc(float v, float& m, float& s) {
    if (v > m) { s = s * fast_exp(m - v) + 1.0f; m = v; }
    else       { s += fast_exp(v - m); }
}

__device__ __forceinline__ void wait_async0() {
#if __has_builtin(__builtin_amdgcn_s_wait_asynccnt)
    __builtin_amdgcn_s_wait_asynccnt(0);
#else
    asm volatile("s_wait_asynccnt 0" ::: "memory");
#endif
}

// Async-DMA one 16x64 f32 feat tile (4KB) into LDS: 8 x b128 per lane.
// Tracked on ASYNCcnt -> overlaps with WMMA/log work of the current step.
__device__ __forceinline__ void stage_feats_async(float* dst /*LDS, rows padded to 65*/,
                                                  const float* gbase,
                                                  int b0, int s, int lane) {
#pragma unroll
    for (int k = 0; k < 8; ++k) {
        int q = lane + k * 32;           // float4 index 0..255
        int r = q >> 4;                  // row 0..15
        int c = (q & 15) * 4;            // col 0..60
        unsigned lds = (unsigned)(uintptr_t)(dst + r * 65 + c);
        int goff = (int)(((size_t)(b0 + r) * (size_t)(S_ * T_) +
                          (size_t)s * T_ + (size_t)c) * sizeof(float));
        asm volatile("global_load_async_to_lds_b128 %0, %1, %2"
                     :: "v"(lds), "v"(goff), "s"(gbase) : "memory");
    }
}

__global__ __launch_bounds__(32)
void crf_fused_kernel(const float* __restrict__ feats,
                      const float* __restrict__ trans,
                      const float* __restrict__ startT,
                      const float* __restrict__ endT,
                      const int*   __restrict__ tags,
                      float*       __restrict__ out)
{
    __shared__ _Float16 expT  [64 * 64];    // exp(trans) f16, [from][to]
    __shared__ float    transF[64 * 65];    // trans f32 for gold path
    __shared__ float    alphaS [16 * 65];   // forward alpha
    __shared__ float    galphaS[16 * 65];   // gold alpha
    __shared__ float    featS[2][16 * 65];  // double-buffered feat tile
    __shared__ float    aM[16];             // per-row max of alpha
    __shared__ float    stS[64], enS[64];
    __shared__ int      curtagS[16], ptagS[16];

    const int lane = threadIdx.x;      // 0..31 (wave32)
    const int g    = lane >> 4;        // lane group 0/1
    const int ln   = lane & 15;
    const int b0   = blockIdx.x * 16;  // 16 batch rows per wave

    // ---- stage transitions: f32 copy + exp() f16 copy ----
    for (int idx = lane; idx < 64 * 64; idx += 32) {
        float t = trans[idx];
        transF[(idx >> 6) * 65 + (idx & 63)] = t;
        expT[idx] = (_Float16)fast_exp(t);
    }
    stS[lane] = startT[lane]; stS[lane + 32] = startT[lane + 32];
    enS[lane] = endT[lane];   enS[lane + 32] = endT[lane + 32];

    // kick off async DMA of the s=0 feat tile while we build B fragments
    stage_feats_async(featS[0], feats, b0, 0, lane);
    __syncthreads();

    // ---- loop-invariant B fragments: exp(trans), 4 N-tiles x 2 K-halves ----
    v16h Bfr[4][2];
#pragma unroll
    for (int nt = 0; nt < 4; ++nt) {
#pragma unroll
        for (int h = 0; h < 2; ++h) {
            v16h b;
#pragma unroll
            for (int e = 0; e < 16; ++e) {
                int k = e + 16 * g + 32 * h;
                int n = ln + 16 * nt;
                b[e] = expT[k * 64 + n];
            }
            Bfr[nt][h] = b;
        }
    }

    if (lane < 16) curtagS[lane] = tags[(size_t)(b0 + lane) * S_];
    wait_async0();             // s=0 tile landed in LDS
    __syncthreads();

    int cnt = 0;
    if (lane < 16) cnt = (curtagS[lane] != 0) ? 1 : 0;

    // ---- s = 0 init ----
    for (int r = 0; r < 16; ++r) {
        int  t0   = curtagS[r];
        bool alln = (t0 == 1);          // unlabeled -> all tags possible
#pragma unroll
        for (int jj = 0; jj < 2; ++jj) {
            int   j  = lane + jj * 32;
            float a0 = stS[j] + featS[0][r * 65 + j];
            alphaS [r * 65 + j] = a0;
            galphaS[r * 65 + j] = (alln || j == t0) ? a0 : NEGV;
        }
    }
    // prefetch s=1 tile asynchronously
    stage_feats_async(featS[1], feats, b0, 1, lane);
    __syncthreads();

    // ================= sequential scan over s =================
    for (int s = 1; s < S_; ++s) {
        const int buf = s & 1;
        const float* fS = featS[buf];

        if (lane < 16) {
            ptagS[lane] = curtagS[lane];
            int ct = tags[(size_t)(b0 + lane) * S_ + s];
            curtagS[lane] = ct;
            cnt += (ct != 0);
        }

        wait_async0();                       // this step's tile is ready
        if (s + 1 < S_)                      // overlap next tile's DMA with compute
            stage_feats_async(featS[buf ^ 1], feats, b0, s + 1, lane);
        __syncthreads();

        // per-row max of alpha: each lane does a 32-elem half, combine via shfl
        float mrow;
        {
            float m = alphaS[ln * 65 + g * 32];
            for (int j = 1; j < 32; ++j) m = fmaxf(m, alphaS[ln * 65 + g * 32 + j]);
            m = fmaxf(m, __shfl_xor(m, 16, 32));
            mrow = m;                        // full max of row ln (both groups)
            if (g == 0) aM[ln] = m;
        }

        // A fragments: exp(alpha - max), 16x32 f16 layout
        v16h a0, a1;
#pragma unroll
        for (int e = 0; e < 16; ++e) {
            int kk = e + g * 16;
            a0[e] = (_Float16)fast_exp(alphaS[ln * 65 + kk]      - mrow);
            a1[e] = (_Float16)fast_exp(alphaS[ln * 65 + kk + 32] - mrow);
        }

        // preload per-lane row info for the branchless epilogue (rows rv+8g)
        float aMr[8];
        float rowon[8];
#pragma unroll
        for (int rv = 0; rv < 8; ++rv) {
            int M = rv + 8 * g;
            aMr[rv]   = aM[M];
            rowon[rv] = (curtagS[M] != 0) ? 1.0f : 0.0f;
        }

        // 16x64x64 exp-space GEMM: 4 N-tiles x 2 K-steps of v_wmma_f32_16x16x32_f16
#pragma unroll
        for (int nt = 0; nt < 4; ++nt) {
            v8f c = {0.f, 0.f, 0.f, 0.f, 0.f, 0.f, 0.f, 0.f};
            c = __builtin_amdgcn_wmma_f32_16x16x32_f16(false, a0, false, Bfr[nt][0],
                                                       (short)0, c, false, false);
            c = __builtin_amdgcn_wmma_f32_16x16x32_f16(false, a1, false, Bfr[nt][1],
                                                       (short)0, c, false, false);
            int N = ln + 16 * nt;
#pragma unroll
            for (int rv = 0; rv < 8; ++rv) {
                int   M    = rv + 8 * g;       // C/D layout: vgpr rv, lane group g
                float oldv = alphaS[M * 65 + N];
                float nv   = fast_log(c[rv]) + aMr[rv] + fS[M * 65 + N];
                alphaS[M * 65 + N] = (rowon[rv] != 0.0f) ? nv : oldv;  // branchless gate
            }
        }
        __syncthreads();

        // ---- gold path (masked score), per row; 2 tag-columns per lane ----
        for (int r = 0; r < 16; ++r) {
            int ct = curtagS[r];
            if (ct == 0) continue;           // masked step: galpha unchanged
            int  pt     = ptagS[r];
            bool allcur = (pt == 1);
            bool allnxt = (ct == 1);
            float new0, new1;
            if (!allcur) {
                // one-hot previous tag: NEG-masked terms underflow exactly -> single path
                float gci = galphaS[r * 65 + pt];
#pragma unroll
                for (int jj = 0; jj < 2; ++jj) {
                    int   j  = lane + jj * 32;
                    bool  nx = allnxt || (j == ct);
                    float fj = nx ? fS[r * 65 + j] : NEGV;
                    float tv = transF[pt * 65 + j];
                    float tm = (nx && tv != 0.0f) ? tv : NEGV;
                    float nv = gci + fj + tm;
                    if (jj == 0) new0 = nv; else new1 = nv;
                }
            } else {
                // unlabeled previous tag: full streaming lse over i
#pragma unroll
                for (int jj = 0; jj < 2; ++jj) {
                    int   j  = lane + jj * 32;
                    bool  nx = allnxt || (j == ct);
                    float fj = nx ? fS[r * 65 + j] : NEGV;
                    float m = -3.0e38f, ssum = 0.0f;
                    for (int i = 0; i < 64; ++i) {
                        float tv = transF[i * 65 + j];
                        float tm = (nx && tv != 0.0f) ? tv : NEGV;
                        lse_acc(galphaS[r * 65 + i] + tm, m, ssum);
                    }
                    float nv = fj + m + fast_log(ssum);
                    if (jj == 0) new0 = nv; else new1 = nv;
                }
            }
            galphaS[r * 65 + lane]      = new0;
            galphaS[r * 65 + lane + 32] = new1;
        }
        __syncthreads();
    }

    // ================= finalize: fwd - gold per row =================
    if (lane < 16) {
        int row = lane;
        float m = -3.0e38f, ssum = 0.0f;
        for (int j = 0; j < 64; ++j) lse_acc(alphaS[row * 65 + j] + enS[j], m, ssum);
        float fwd = m + fast_log(ssum);

        int p = cnt - 1; if (p < 0) p = 0;
        int  tl   = tags[(size_t)(b0 + row) * S_ + p];  // tag at last valid position
        bool alle = (tl == 1);
        float mg = -3.0e38f, sg = 0.0f;
        for (int j = 0; j < 64; ++j) {
            float e  = enS[j];
            float et = ((alle || j == tl) && e != 0.0f) ? e : NEGV;
            lse_acc(galphaS[row * 65 + j] + et, mg, sg);
        }
        float gold = mg + fast_log(sg);
        atomicAdd(out, fwd - gold);
    }
}

extern "C" void kernel_launch(void* const* d_in, const int* in_sizes, int n_in,
                              void* d_out, int out_size, void* d_ws, size_t ws_size,
                              hipStream_t stream) {
    (void)in_sizes; (void)n_in; (void)out_size; (void)d_ws; (void)ws_size;
    const float* feats  = (const float*)d_in[0];
    const float* trans  = (const float*)d_in[1];
    const float* startT = (const float*)d_in[2];
    const float* endT   = (const float*)d_in[3];
    const int*   tags   = (const int*)d_in[4];
    float* out = (float*)d_out;

    hipMemsetAsync(out, 0, sizeof(float), stream);
    crf_fused_kernel<<<B_ / 16, 32, 0, stream>>>(feats, trans, startT, endT, tags, out);
}
